// SeqDecoderWrapper_91104846282852
// MI455X (gfx1250) — matrix operand order repaired and checked by reading, not compiled, not run on hardware
//
#include <hip/hip_runtime.h>
#include <hip/hip_bf16.h>
#include <cstdint>
#include <cstddef>

// Problem dims (match reference)
#define B_     1024
#define IN_    64
#define H_     1024
#define OUT_   108
#define OUTP_  112     // OUT padded to 7*16 for WMMA N tiles
#define STEPS_ 64
#define G4H_   4096    // 4*H

#define KSTAGE   4                         // k-steps (of 32) per LDS stage
#define NSTAGES  (H_ / (KSTAGE * 32))      // 8
#define FRAG_ELEMS 512                     // one 32x16 bf16 B-frag: 32 lanes * 16 elems

typedef __bf16 bf16_t;
typedef __attribute__((ext_vector_type(16))) __bf16 v16bf;
typedef __attribute__((ext_vector_type(8)))  __bf16 v8bf;
typedef __attribute__((ext_vector_type(8)))  float  v8f;
typedef int vint4 __attribute__((vector_size(16)));

__device__ __forceinline__ float sigm_f(float x)  { return 1.0f / (1.0f + __expf(-x)); }
__device__ __forceinline__ float tanh_f(float x)  { return 1.0f - 2.0f / (__expf(2.0f * x) + 1.0f); }

// ---------------------------------------------------------------------------
// Async global->LDS staging (CDNA5). Guarded: falls back to load+ds_store.
// ---------------------------------------------------------------------------
#if __has_builtin(__builtin_amdgcn_global_load_async_to_lds_b128) && \
    __has_builtin(__builtin_amdgcn_s_wait_asynccnt)
#define USE_ASYNC_LDS 1
#else
#define USE_ASYNC_LDS 0
#endif

__device__ __forceinline__ void async_copy16(const bf16_t* src, bf16_t* dst_lds) {
#if USE_ASYNC_LDS
  // Builtin signature: (v4i addrspace(1)* src, v4i addrspace(3)* dst, imm off, imm cpol).
  // Cast through integers: 64-bit for global, low 32 bits of a generic LDS
  // pointer are the LDS byte offset (ISA §10.2 aperture rules).
  __builtin_amdgcn_global_load_async_to_lds_b128(
      (__attribute__((address_space(1))) vint4*)(uintptr_t)src,
      (__attribute__((address_space(3))) vint4*)(uint32_t)(uintptr_t)dst_lds,
      /*offset=*/0, /*cpol=*/0);
#else
  *(v8bf*)dst_lds = *(const v8bf*)src;   // synchronous fallback, same layout
#endif
}

#if USE_ASYNC_LDS
#define WAIT_ASYNC(n) __builtin_amdgcn_s_wait_asynccnt(n)
#else
#define WAIT_ASYNC(n) ((void)0)
#endif

// ---------------------------------------------------------------------------
// WMMA fragment helpers (bf16 16x16x32), in this file's own contiguous-K order (not AMD's table order)
// A 16x32 (MxK): lane holds row M=(lane&15); half=lane>>4 ->
//   half=0: K in [0..15]; half=1: K in [16..31] (contiguous, matches B)
// B 32x16 (KxN): lane holds col N=(lane&15); lanes 0-15: K=0..15,
//   lanes 16-31: K=16..31 (16 contiguous K per lane = 32B).
// ---------------------------------------------------------------------------
__device__ __forceinline__ v16bf load_fragA(const bf16_t* __restrict__ base,
                                            int m0, int k0, int ld, int lane) {
  const int half = lane >> 4, lc = lane & 15;
  const bf16_t* p = base + (size_t)(m0 + lc) * ld + (size_t)(k0 + half * 16);
  v8bf lo = *(const v8bf*)(p);
  v8bf hi = *(const v8bf*)(p + 8);
  return __builtin_shufflevector(lo, hi, 0,1,2,3,4,5,6,7,8,9,10,11,12,13,14,15);
}

__device__ __forceinline__ v16bf load_fragB(const bf16_t* __restrict__ base,
                                            int n0, int k0, int ld, int lane) {
  const int half = lane >> 4, lc = lane & 15;
  const bf16_t* p = base + (size_t)(n0 + lc) * ld + (size_t)(k0 + half * 16);
  v8bf lo = *(const v8bf*)(p);
  v8bf hi = *(const v8bf*)(p + 8);
  return __builtin_shufflevector(lo, hi, 0,1,2,3,4,5,6,7,8,9,10,11,12,13,14,15);
}

// B-frag read from LDS: frags stored pre-swizzled, 32B contiguous per lane.
__device__ __forceinline__ v16bf load_fragB_lds(const bf16_t* lds, int buf,
                                                int g, int t, int lane) {
  const bf16_t* p = lds + (((buf * 4 + g) * KSTAGE + t) * FRAG_ELEMS + lane * 16);
  v8bf lo = *(const v8bf*)(p);
  v8bf hi = *(const v8bf*)(p + 8);
  return __builtin_shufflevector(lo, hi, 0,1,2,3,4,5,6,7,8,9,10,11,12,13,14,15);
}

__device__ __forceinline__ v8f wmma_bf16(v16bf a, v16bf b, v8f c) {
  return __builtin_amdgcn_wmma_f32_16x16x32_bf16(false, a, false, b, (short)0, c,
                                                 false, false);
}

// ---------------------------------------------------------------------------
// Stage 4 gate B-tiles for KSTAGE k-steps into LDS (pre-swizzled frag layout).
// 32 b128 transfers per stage; 8 waves issue 4 each. Per lane: 16B chunk of
// W row (g*H + n0 + (lane&15)), K chunk selected by (lane>>4, part).
// ---------------------------------------------------------------------------
__device__ __forceinline__ void issue_stageB(const bf16_t* __restrict__ W, int n0,
                                             bf16_t* lds, int buf, int k0,
                                             int wavein, int lane) {
  const int half = lane >> 4, lc = lane & 15;
#pragma unroll
  for (int q = 0; q < 4; ++q) {
    const int id = wavein * 4 + q;   // 0..31
    const int p  = id & 1;           // 16B part within lane's 32B
    const int gt = id >> 1;          // 0..15
    const int g  = gt >> 2;          // gate 0..3
    const int t  = gt & 3;           // k-step within stage (KSTAGE=4)
    const bf16_t* src =
        W + (size_t)(g * H_ + n0 + lc) * H_ + (size_t)(k0 + t * 32 + half * 16 + p * 8);
    bf16_t* dst = lds + (((buf * 4 + g) * KSTAGE + t) * FRAG_ELEMS + lane * 16 + p * 8);
    async_copy16(src, dst);
  }
}

__device__ __forceinline__ void compute_stage(v8f acc[2][4],
                                              const bf16_t* __restrict__ A,
                                              const bf16_t* lds, int buf,
                                              int m0, int k0, int lane) {
#pragma unroll
  for (int t = 0; t < KSTAGE; ++t) {
    const int k = k0 + t * 32;
    v16bf a0 = load_fragA(A, m0,      k, H_, lane);
    v16bf a1 = load_fragA(A, m0 + 16, k, H_, lane);
#pragma unroll
    for (int g = 0; g < 4; ++g) {
      v16bf b = load_fragB_lds(lds, buf, g, t, lane);
      acc[0][g] = wmma_bf16(a0, b, acc[0][g]);
      acc[1][g] = wmma_bf16(a1, b, acc[1][g]);
    }
  }
}

// Double-buffered gates GEMM: acc += A[B,H] @ W[4H,H]^T (four gate columns n0..n0+15)
__device__ __forceinline__ void gemm_lds(v8f acc[2][4],
                                         const bf16_t* __restrict__ A,
                                         const bf16_t* __restrict__ W,
                                         bf16_t* lds, int m0, int n0,
                                         int wavein, int lane) {
  issue_stageB(W, n0, lds, 0, 0, wavein, lane);
#pragma unroll 1
  for (int st = 0; st < NSTAGES; ++st) {
    const int buf = st & 1;
    if (st + 1 < NSTAGES) {
      issue_stageB(W, n0, lds, buf ^ 1, (st + 1) * KSTAGE * 32, wavein, lane);
      WAIT_ASYNC(4);   // our 4 stage-(st) copies done; stage-(st+1) may fly
    } else {
      WAIT_ASYNC(0);
    }
    __syncthreads();   // all waves' stage-(st) data visible in LDS
    compute_stage(acc, A, lds, buf, m0, st * KSTAGE * 32, lane);
    __syncthreads();   // safe to overwrite buf next iteration
  }
}

// ---------------------------------------------------------------------------
// Fused LSTM layer: gates GEMM (bf16 WMMA, f32 acc, LDS-staged weights) + cell.
// Block = 8 waves sharing one hidden tile: 256 batch rows x 16 hidden cols.
// Wave tile: 32(batch) x 16(hidden) x 4 gates = 8 accumulators.
// ---------------------------------------------------------------------------
template <bool LAYER1>
__global__ void lstm_layer_kernel(const bf16_t* __restrict__ Aprev,  // h_prev [B,H]
                                  const bf16_t* __restrict__ Whh,    // [4H,H]
                                  const bf16_t* __restrict__ Ain,    // h0_new [B,H] (layer1)
                                  const bf16_t* __restrict__ Wih,    // [4H,H]  (layer1)
                                  const float*  __restrict__ initv,  // gx0 [B,4H] or bias1 [4H]
                                  float*        __restrict__ cstate, // [B,H] (in-place)
                                  bf16_t*       __restrict__ hout)   // [B,H]
{
  __shared__ bf16_t ldsB[2 * 4 * KSTAGE * FRAG_ELEMS];   // 32 KB

  const int tid    = threadIdx.x;
  const int lane   = tid & 31;
  const int wavein = tid >> 5;
  const int m0     = (blockIdx.x >> 6) * 256 + wavein * 32;  // B/256 = 4 groups
  const int n0     = (blockIdx.x & 63) * 16;                 // H/16  = 64 tiles
  const int half   = lane >> 4, lc = lane & 15;

  v8f acc[2][4];
  if (LAYER1) {
#pragma unroll
    for (int g = 0; g < 4; ++g) {
      const float bv = initv[g * H_ + n0 + lc];
#pragma unroll
      for (int r = 0; r < 8; ++r) { acc[0][g][r] = bv; acc[1][g][r] = bv; }
    }
  } else {
#pragma unroll
    for (int mi = 0; mi < 2; ++mi)
#pragma unroll
      for (int g = 0; g < 4; ++g)
#pragma unroll
        for (int r = 0; r < 8; ++r)
          acc[mi][g][r] =
              initv[(size_t)(m0 + mi * 16 + half * 8 + r) * G4H_ + g * H_ + n0 + lc];
  }

  gemm_lds(acc, Aprev, Whh, ldsB, m0, n0, wavein, lane);
  if (LAYER1) gemm_lds(acc, Ain, Wih, ldsB, m0, n0, wavein, lane);

  // Fused LSTM cell epilogue. D layout: VGPR r -> row (m0+mi*16+half*8+r), col n0+lc.
#pragma unroll
  for (int mi = 0; mi < 2; ++mi) {
#pragma unroll
    for (int r = 0; r < 8; ++r) {
      const int    brow = m0 + mi * 16 + half * 8 + r;
      const size_t idx  = (size_t)brow * H_ + n0 + lc;
      const float iv = sigm_f(acc[mi][0][r]);
      const float fv = sigm_f(acc[mi][1][r]);
      const float gv = tanh_f(acc[mi][2][r]);
      const float ov = sigm_f(acc[mi][3][r]);
      const float cn = fv * cstate[idx] + iv * gv;
      cstate[idx] = cn;
      hout[idx]   = (bf16_t)(ov * tanh_f(cn));
    }
  }
}

// ---------------------------------------------------------------------------
// gx0 = x @ W_ih0^T + b_ih0 + b_hh0 (one-time; K=64 -> 2 WMMA steps per tile)
// ---------------------------------------------------------------------------
__global__ void gx0_kernel(const bf16_t* __restrict__ Xbf,    // [B,64]
                           const bf16_t* __restrict__ Wih0,   // [4H,64]
                           const float*  __restrict__ b_ih0,
                           const float*  __restrict__ b_hh0,
                           float*        __restrict__ gx0)    // [B,4H]
{
  const int tid  = threadIdx.x;
  const int lane = tid & 31;
  const int wave = blockIdx.x * (blockDim.x >> 5) + (tid >> 5);
  const int m0   = (wave >> 8) * 16;    // B/16  = 64
  const int n0   = (wave & 255) * 16;   // 4H/16 = 256
  const int half = lane >> 4, lc = lane & 15;

  v8f acc;
  const float bv = b_ih0[n0 + lc] + b_hh0[n0 + lc];
#pragma unroll
  for (int r = 0; r < 8; ++r) acc[r] = bv;

#pragma unroll
  for (int k = 0; k < IN_; k += 32) {
    v16bf a = load_fragA(Xbf,  m0, k, IN_, lane);
    v16bf b = load_fragB(Wih0, n0, k, IN_, lane);
    acc = wmma_bf16(a, b, acc);
  }
#pragma unroll
  for (int r = 0; r < 8; ++r)
    gx0[(size_t)(m0 + half * 8 + r) * G4H_ + n0 + lc] = acc[r];
}

// ---------------------------------------------------------------------------
// Per-step FC: out[b, o, s] = h1[b,:] . W_fc[o,:] + b_fc[o]
// ---------------------------------------------------------------------------
__global__ void fc_kernel(const bf16_t* __restrict__ h1,    // [B,H]
                          const bf16_t* __restrict__ Wfc,   // [112,H] (zero padded)
                          const float*  __restrict__ bfc,   // [108]
                          float*        __restrict__ out,   // [B,108,64]
                          int s)
{
  const int tid  = threadIdx.x;
  const int lane = tid & 31;
  const int wave = blockIdx.x * (blockDim.x >> 5) + (tid >> 5);
  const int m0   = (wave / 7) * 16;  // B/16 = 64 strips
  const int n0   = (wave % 7) * 16;  // 7 N tiles (112 cols)
  const int half = lane >> 4, lc = lane & 15;
  const int ocol = n0 + lc;

  v8f acc;
  const float bv = (ocol < OUT_) ? bfc[ocol] : 0.0f;
#pragma unroll
  for (int r = 0; r < 8; ++r) acc[r] = bv;

  for (int k = 0; k < H_; k += 32) {
    v16bf a = load_fragA(h1,  m0, k, H_, lane);
    v16bf b = load_fragB(Wfc, n0, k, H_, lane);
    acc = wmma_bf16(a, b, acc);
  }
  if (ocol < OUT_) {
#pragma unroll
    for (int r = 0; r < 8; ++r) {
      const int brow = m0 + half * 8 + r;
      out[((size_t)brow * OUT_ + ocol) * STEPS_ + s] = acc[r];
    }
  }
}

// ---------------------------------------------------------------------------
// Prep utilities
// ---------------------------------------------------------------------------
__global__ void cvt_f32_bf16_kernel(const float* __restrict__ src,
                                    bf16_t* __restrict__ dst, int n) {
  int i = blockIdx.x * blockDim.x + threadIdx.x;
  if (i < n) dst[i] = (bf16_t)src[i];
}

__global__ void pad_wfc_kernel(const float* __restrict__ src,   // [108,1024]
                               bf16_t* __restrict__ dst) {      // [112,1024]
  int i = blockIdx.x * blockDim.x + threadIdx.x;
  if (i < OUTP_ * H_) {
    int row = i / H_;
    dst[i] = (row < OUT_) ? (bf16_t)src[i] : (bf16_t)0.0f;
  }
}

__global__ void add_bias_kernel(const float* __restrict__ a,
                                const float* __restrict__ b,
                                float* __restrict__ dst, int n) {
  int i = blockIdx.x * blockDim.x + threadIdx.x;
  if (i < n) dst[i] = a[i] + b[i];
}

// ---------------------------------------------------------------------------
extern "C" void kernel_launch(void* const* d_in, const int* in_sizes, int n_in,
                              void* d_out, int out_size, void* d_ws, size_t ws_size,
                              hipStream_t stream) {
  (void)in_sizes; (void)n_in; (void)out_size; (void)ws_size;

  const float* frame0 = (const float*)d_in[0];   // [B,1,64]
  const float* h0in   = (const float*)d_in[1];   // [2,B,H]
  const float* c0in   = (const float*)d_in[2];   // [2,B,H]
  const float* W_ih0  = (const float*)d_in[3];   // [4H,64]
  const float* W_hh0  = (const float*)d_in[4];   // [4H,H]
  const float* b_ih0  = (const float*)d_in[5];
  const float* b_hh0  = (const float*)d_in[6];
  const float* W_ih1  = (const float*)d_in[7];   // [4H,H]
  const float* W_hh1  = (const float*)d_in[8];   // [4H,H]
  const float* b_ih1  = (const float*)d_in[9];
  const float* b_hh1  = (const float*)d_in[10];
  const float* W_fc   = (const float*)d_in[11];  // [108,H]
  const float* b_fc   = (const float*)d_in[12];
  float* out = (float*)d_out;

  // Workspace carve-up (256B aligned)
  char* ws = (char*)d_ws;
  size_t off = 0;
  auto carve = [&](size_t bytes) -> char* {
    char* p = ws + off;
    off = (off + bytes + 255) & ~(size_t)255;
    return p;
  };
  bf16_t* Whh0_bf = (bf16_t*)carve((size_t)G4H_ * H_  * 2);
  bf16_t* Wih1_bf = (bf16_t*)carve((size_t)G4H_ * H_  * 2);
  bf16_t* Whh1_bf = (bf16_t*)carve((size_t)G4H_ * H_  * 2);
  bf16_t* Wfc_bf  = (bf16_t*)carve((size_t)OUTP_ * H_ * 2);
  bf16_t* Wih0_bf = (bf16_t*)carve((size_t)G4H_ * IN_ * 2);
  bf16_t* X_bf    = (bf16_t*)carve((size_t)B_ * IN_   * 2);
  float*  gx0     = (float*)carve((size_t)B_ * G4H_   * 4);
  float*  bias1   = (float*)carve((size_t)G4H_        * 4);
  bf16_t* h0bf[2] = { (bf16_t*)carve((size_t)B_ * H_ * 2),
                      (bf16_t*)carve((size_t)B_ * H_ * 2) };
  bf16_t* h1bf[2] = { (bf16_t*)carve((size_t)B_ * H_ * 2),
                      (bf16_t*)carve((size_t)B_ * H_ * 2) };
  float*  c0ws    = (float*)carve((size_t)B_ * H_ * 4);
  float*  c1ws    = (float*)carve((size_t)B_ * H_ * 4);

  const int T = 256;
  auto blocks = [](size_t n, int t) { return (unsigned)((n + t - 1) / t); };

  // ---- Prep: bf16 weight/state images, combined biases, gx0 ----
  cvt_f32_bf16_kernel<<<blocks((size_t)G4H_ * H_, T), T, 0, stream>>>(W_hh0, Whh0_bf, G4H_ * H_);
  cvt_f32_bf16_kernel<<<blocks((size_t)G4H_ * H_, T), T, 0, stream>>>(W_ih1, Wih1_bf, G4H_ * H_);
  cvt_f32_bf16_kernel<<<blocks((size_t)G4H_ * H_, T), T, 0, stream>>>(W_hh1, Whh1_bf, G4H_ * H_);
  cvt_f32_bf16_kernel<<<blocks((size_t)G4H_ * IN_, T), T, 0, stream>>>(W_ih0, Wih0_bf, G4H_ * IN_);
  cvt_f32_bf16_kernel<<<blocks((size_t)B_ * IN_, T), T, 0, stream>>>(frame0, X_bf, B_ * IN_);
  pad_wfc_kernel<<<blocks((size_t)OUTP_ * H_, T), T, 0, stream>>>(W_fc, Wfc_bf);
  add_bias_kernel<<<blocks(G4H_, T), T, 0, stream>>>(b_ih1, b_hh1, bias1, G4H_);
  cvt_f32_bf16_kernel<<<blocks((size_t)B_ * H_, T), T, 0, stream>>>(h0in, h0bf[0], B_ * H_);
  cvt_f32_bf16_kernel<<<blocks((size_t)B_ * H_, T), T, 0, stream>>>(h0in + (size_t)B_ * H_, h1bf[0], B_ * H_);
  (void)hipMemcpyAsync(c0ws, c0in, (size_t)B_ * H_ * 4, hipMemcpyDeviceToDevice, stream);
  (void)hipMemcpyAsync(c1ws, c0in + (size_t)B_ * H_, (size_t)B_ * H_ * 4,
                       hipMemcpyDeviceToDevice, stream);

  // gx0: (B/16)*(4H/16) = 16384 waves -> 2048 blocks of 8 waves
  gx0_kernel<<<2048, T, 0, stream>>>(X_bf, Wih0_bf, b_ih0, b_hh0, gx0);

  // ---- Recurrence: 64 steps, stream order provides the dependency chain ----
  // lstm layer grids: (B/256)*(H/16) = 256 blocks of 8 waves
  for (int s = 0; s < STEPS_; ++s) {
    const int pi = s & 1, po = pi ^ 1;
    lstm_layer_kernel<false><<<256, T, 0, stream>>>(
        h0bf[pi], Whh0_bf, nullptr, nullptr, gx0, c0ws, h0bf[po]);
    lstm_layer_kernel<true><<<256, T, 0, stream>>>(
        h1bf[pi], Whh1_bf, h0bf[po], Wih1_bf, bias1, c1ws, h1bf[po]);
    // FC: (B/16)*7 = 448 waves -> 56 blocks
    fc_kernel<<<56, T, 0, stream>>>(h1bf[po], Wfc_bf, b_fc, out, s);
  }
}
